// TreeEmbeddingModel_13005160972556
// MI455X (gfx1250) — matrix operand order repaired and checked by reading, not compiled, not run on hardware
//
#include <hip/hip_runtime.h>

typedef _Float16 f16;
typedef __attribute__((ext_vector_type(16))) _Float16 v16h;
typedef __attribute__((ext_vector_type(4)))  _Float16 v4h;
typedef __attribute__((ext_vector_type(8)))  float    v8f;

#define BM 128
#define BN 64
#define KT 64
#define AST 72   // LDS row stride in halves: 144B = 36 banks -> conflict-free b128 reads
#define BST 72

// ---------------- gather: labels[b,:] = class_embedding[b, idx[b], :] ----------------
__global__ void gather_labels_kernel(const float* __restrict__ cls,
                                     const int* __restrict__ idx,
                                     float* __restrict__ out,
                                     int C, int H, int total4) {
  int t = blockIdx.x * blockDim.x + threadIdx.x;
  if (t >= total4) return;
  int h4pr = H >> 2;
  int b = t / h4pr;
  int p = t - b * h4pr;
  int c = idx[b];
  const float4* src = (const float4*)(cls + ((size_t)b * C + c) * H);
  ((float4*)out)[(size_t)b * h4pr + p] = src[p];
}

// ---------------- weights f32 -> f16 (once; reused S steps x 2 GEMMs) ----------------
__global__ void convert_w_kernel(const float* __restrict__ wh,
                                 const float* __restrict__ wg,
                                 f16* __restrict__ whf,
                                 f16* __restrict__ wgf,
                                 size_t total4) {
  size_t t = (size_t)blockIdx.x * blockDim.x + threadIdx.x;
  if (t >= total4) return;
  float4 a = ((const float4*)wh)[t];
  float4 b = ((const float4*)wg)[t];
  v4h ah = { (f16)a.x, (f16)a.y, (f16)a.z, (f16)a.w };
  v4h gh = { (f16)b.x, (f16)b.y, (f16)b.z, (f16)b.w };
  ((v4h*)whf)[t] = ah;
  ((v4h*)wgf)[t] = gh;
}

// ---------------- one merge step: rout = mask ? tanh(xWh+bh)*sigmoid(xWg+bg) : rin ----------------
__launch_bounds__(256)
__attribute__((amdgpu_waves_per_eu(4)))
__global__ void step_gemm_kernel(const float* __restrict__ op,
                                 const float* __restrict__ lft,
                                 const float* __restrict__ rin,
                                 const f16* __restrict__ whf,
                                 const f16* __restrict__ wgf,
                                 const float* __restrict__ bias_h,
                                 const float* __restrict__ bias_g,
                                 const unsigned char* __restrict__ msk,
                                 float* __restrict__ rout,
                                 int H) {
  const int K3 = 3 * H;
  __shared__ __align__(16) f16 As [2][BM][AST];
  __shared__ __align__(16) f16 Bsh[2][BN][BST];
  __shared__ __align__(16) f16 Bsg[2][BN][BST];

  const int tid  = threadIdx.x;
  const int lane = tid & 31;
  const int wv   = tid >> 5;                 // 0..7 waves
  const int Mblk = blockIdx.y * BM;
  const int Nblk = blockIdx.x * BN;

  // staging coords
  const int ar0 = tid >> 4;                  // 0..15
  const int ac4 = (tid & 15) * 4;            // 0..60 (floats/halves)
  const int bn0 = tid >> 2;                  // 0..63
  const int bc8 = (tid & 3) * 8;             // halves (16B chunks)

  // fragment coords (wave32 16-bit A / B with k = 16*half + s, f32 C-D layout)
  const int arow = lane & 15;
  const int asel = (lane >> 4) * 8;          // A: K 0-15 vs 16-31 (read at 2*asel)
  const int bcol = lane & 15;
  const int bsel = (lane >> 4) * 16;         // B: K 0-15 vs 16-31
  const int rsel = (lane >> 4) * 8;          // C/D: rows +0 vs +8

  v8f acch[4] = {};
  v8f accg[4] = {};

  // A tile: f32 source -> f16 LDS; batch loads first so waits step down
  auto stageA = [&](int buf, int kb) {
    const float* src; int kl;
    if (kb < H)          { src = op;  kl = kb;         }
    else if (kb < 2 * H) { src = lft; kl = kb - H;     }
    else                 { src = rin; kl = kb - 2 * H; }
    float4 va[8];
#pragma unroll
    for (int p = 0; p < 8; ++p)
      va[p] = *(const float4*)(src + (size_t)(Mblk + p * 16 + ar0) * H + kl + ac4);
#pragma unroll
    for (int p = 0; p < 8; ++p) {
      v4h hv = { (f16)va[p].x, (f16)va[p].y, (f16)va[p].z, (f16)va[p].w };
      *(v4h*)&As[buf][p * 16 + ar0][ac4] = hv;
    }
  };

  // B tiles: already f16 -> async global->LDS copy (no VGPR round-trip, ASYNCcnt)
  auto stageB = [&](int buf, int kb) {
#pragma unroll
    for (int kk = 0; kk < KT; kk += 32) {
      size_t wo = (size_t)(Nblk + bn0) * K3 + kb + kk + bc8;
      unsigned ldh = (unsigned)(size_t)&Bsh[buf][bn0][kk + bc8];
      unsigned ldg = (unsigned)(size_t)&Bsg[buf][bn0][kk + bc8];
      asm volatile("global_load_async_to_lds_b128 %0, %1, off"
                   :: "v"(ldh), "v"((unsigned long long)(size_t)(whf + wo))
                   : "memory");
      asm volatile("global_load_async_to_lds_b128 %0, %1, off"
                   :: "v"(ldg), "v"((unsigned long long)(size_t)(wgf + wo))
                   : "memory");
    }
  };

  stageA(0, 0);
  stageB(0, 0);
  asm volatile("s_wait_asynccnt 0x0" ::: "memory");
  __syncthreads();

  const int nk = K3 / KT;
  for (int i = 0; i < nk; ++i) {
    const int cur = i & 1;
    if (i + 1 < nk) { stageA(cur ^ 1, (i + 1) * KT); stageB(cur ^ 1, (i + 1) * KT); }

#pragma unroll
    for (int ks = 0; ks < 2; ++ks) {
      const f16* ap = &As[cur][wv * 16 + arow][ks * 32];
      v16h afrag;
      ((float4*)&afrag)[0] = *(const float4*)(ap + 2 * asel);
      ((float4*)&afrag)[1] = *(const float4*)(ap + 2 * asel + 8);

      const int ko = ks * 32 + bsel;
      // All 8 B fragments in distinct, simultaneously-live registers:
      // 16 back-to-back ds_load_b128, then 8 WMMAs -> descending dscnt waits.
      v16h bf[8];
#pragma unroll
      for (int t = 0; t < 4; ++t) {
        const f16* nh = &Bsh[cur][t * 16 + bcol][ko];
        ((float4*)&bf[t])[0] = *(const float4*)nh;
        ((float4*)&bf[t])[1] = *(const float4*)(nh + 8);
      }
#pragma unroll
      for (int t = 0; t < 4; ++t) {
        const f16* ng = &Bsg[cur][t * 16 + bcol][ko];
        ((float4*)&bf[4 + t])[0] = *(const float4*)ng;
        ((float4*)&bf[4 + t])[1] = *(const float4*)(ng + 8);
      }
      // Pin the schedule: loads stay above, WMMAs below.
      __builtin_amdgcn_sched_barrier(0);
#pragma unroll
      for (int t = 0; t < 4; ++t)
        acch[t] = __builtin_amdgcn_wmma_f32_16x16x32_f16(
            false, afrag, false, bf[t], (short)0, acch[t], false, false);
#pragma unroll
      for (int t = 0; t < 4; ++t)
        accg[t] = __builtin_amdgcn_wmma_f32_16x16x32_f16(
            false, afrag, false, bf[4 + t], (short)0, accg[t], false, false);
    }
    asm volatile("s_wait_asynccnt 0x0" ::: "memory");
    __syncthreads();
  }

  // epilogue: bias + tanh*sigmoid (fast exp forms) + masked select
  const int Mw = Mblk + wv * 16;
#pragma unroll
  for (int r = 0; r < 8; ++r) {
    int m = Mw + r + rsel;
    bool mk = msk[m] != 0;
#pragma unroll
    for (int t = 0; t < 4; ++t) {
      int n = Nblk + t * 16 + (lane & 15);
      float hraw = acch[t][r] + bias_h[n];
      float e2   = __expf(2.0f * hraw);
      float th   = 1.0f - 2.0f / (e2 + 1.0f);
      float gv   = accg[t][r] + bias_g[n];
      float sg   = 1.0f / (1.0f + __expf(-gv));
      size_t off = (size_t)m * H + n;
      rout[off] = mk ? (th * sg) : rin[off];
    }
  }
}

extern "C" void kernel_launch(void* const* d_in, const int* in_sizes, int n_in,
                              void* d_out, int out_size, void* d_ws, size_t ws_size,
                              hipStream_t stream) {
  (void)n_in; (void)out_size; (void)ws_size;
  const float*         cls = (const float*)d_in[0];
  const int*           idx = (const int*)d_in[1];
  const float*         lft = (const float*)d_in[2];
  const float*         op  = (const float*)d_in[3];
  const unsigned char* msk = (const unsigned char*)d_in[4];
  const float*         wh  = (const float*)d_in[5];
  const float*         bh  = (const float*)d_in[6];
  const float*         wg  = (const float*)d_in[7];
  const float*         bg  = (const float*)d_in[8];

  const int H  = in_sizes[6];            // hidden_b has H elements
  const int B  = in_sizes[1];            // embed_node_index has B elements
  const int S  = in_sizes[4] / B;        // step_mask is S*B
  const int C  = in_sizes[0] / (B * H);  // class_embedding is B*C*H
  const int K3 = 3 * H;

  // workspace: [right ping buffer f32 B*H][Wh f16 H*3H][Wg f16 H*3H]
  float* rping = (float*)d_ws;
  f16*   whf   = (f16*)((char*)d_ws + (size_t)B * H * sizeof(float));
  f16*   wgf   = whf + (size_t)H * K3;
  float* rbuf[2] = { (float*)d_out, rping };

  const int start = S & 1;  // parity so the final step writes d_out

  {
    int total4 = (B * H) / 4;
    gather_labels_kernel<<<(total4 + 255) / 256, 256, 0, stream>>>(
        cls, idx, rbuf[start], C, H, total4);
  }
  {
    size_t total4 = (size_t)H * K3 / 4;
    convert_w_kernel<<<(unsigned)((total4 + 255) / 256), 256, 0, stream>>>(
        wh, wg, whf, wgf, total4);
  }
  for (int s = 0; s < S; ++s) {
    const float* rin  = rbuf[(start + s) & 1];
    float*       rout = rbuf[(start + s + 1) & 1];
    dim3 grid(H / BN, B / BM);
    step_gemm_kernel<<<grid, 256, 0, stream>>>(
        op + (size_t)s * B * H, lft + (size_t)s * B * H, rin,
        whf, wgf, bh, bg, msk + (size_t)s * B, rout, H);
  }
}
